// MultiheadSelfAttention_66348654789218
// MI455X (gfx1250) — matrix operand order repaired and checked by reading, not compiled, not run on hardware
//
#include <hip/hip_runtime.h>
#include <math.h>
#include <cstdint>

typedef __attribute__((ext_vector_type(16))) _Float16 v16h;
typedef __attribute__((ext_vector_type(8)))  float    v8f;

#define RS 40            // LDS row stride in halves (80B: 16B-aligned, bank-spread)
#define TILE_BYTES (128 * RS * 2)

// ---------------------------------------------------------------------------
// CDNA5 async global->LDS copy (ASYNCcnt-tracked), inline asm
// ---------------------------------------------------------------------------
__device__ __forceinline__ void async_b128(uint32_t lds, const void* g) {
  asm volatile("global_load_async_to_lds_b128 %0, %1, off"
               :
               : "v"(lds), "v"(g)
               : "memory");
}
__device__ __forceinline__ void wait_async0() {
  asm volatile("s_wait_asynccnt 0" ::: "memory");
}

// ---------------------------------------------------------------------------
// fp32 -> fp16 conversion (one-time, workspace)
// ---------------------------------------------------------------------------
__global__ void f32_to_f16_kernel(const float* __restrict__ in,
                                  _Float16* __restrict__ out, int n) {
  int i = blockIdx.x * blockDim.x + threadIdx.x;
  if (i < n) out[i] = (_Float16)in[i];
}

// transpose + convert: out[c*R + r] = in[r*C + c]   (for Wo^T)
__global__ void transpose_f32_to_f16_kernel(const float* __restrict__ in,
                                            _Float16* __restrict__ out,
                                            int R, int C) {
  int r = blockIdx.y;
  int c = blockIdx.x * blockDim.x + threadIdx.x;
  if (c < C) out[(size_t)c * R + r] = (_Float16)in[(size_t)r * C + c];
}

// ---------------------------------------------------------------------------
// WMMA GEMM:  C[M,N] = A[M,K] * B[N,K]^T   (both operands row-major, K contig)
// Output f32 (Cf) or f16 (Ch) -- exactly one non-null.
// causalSkip : skip tiles entirely above the diagonal (score GEMM).
// aCausal    : A is causal-lower-triangular (attn) -> trim K loop at m0+128.
//
// Block: 256 threads = 8 waves (2x4); block tile 128x128; wave tile 64x32
// (4x2 WMMA accumulators). Double-buffered LDS, async global->LDS staging,
// one barrier per K-step; async fill of next buffer overlaps current WMMAs.
// ---------------------------------------------------------------------------
__global__ __launch_bounds__(256)
void gemm_wmma_kernel(const _Float16* __restrict__ A, int lda,
                      const _Float16* __restrict__ Bm, int ldb,
                      float* __restrict__ Cf, _Float16* __restrict__ Ch, int ldc,
                      int K, int causalSkip, int aCausal) {
  __shared__ alignas(16) _Float16 As[2][128 * RS];
  __shared__ alignas(16) _Float16 Bs[2][128 * RS];

  const int m0 = blockIdx.y * 128;
  const int n0 = blockIdx.x * 128;
  if (causalSkip && n0 >= m0 + 128) return;   // fully masked tile

  const int t    = threadIdx.x;
  const int lane = t & 31;
  const int wave = t >> 5;
  const int wr   = wave >> 2;    // 0..1 : wave row (64 rows each)
  const int wc   = wave & 3;     // 0..3 : wave col (32 cols each)
  const int lm   = lane & 15;
  const int lh   = lane >> 4;

  // staging decomposition: each thread DMAs 32B of A and 32B of B per K-step
  const int rowS = t >> 1;       // 0..127
  const int chS  = t & 1;        // 16-half chunk
  const uint32_t ldsA0 = (uint32_t)(uintptr_t)(&As[0][0]);
  const uint32_t ldsB0 = (uint32_t)(uintptr_t)(&Bs[0][0]);
  const uint32_t sOff  = (uint32_t)(rowS * RS + chS * 16) * 2;

  v8f acc[4][2] = {};

  const int Kend  = aCausal ? ((m0 + 128 < K) ? (m0 + 128) : K) : K;
  const int nIter = Kend >> 5;

  auto stage = [&](int buf, int kk) {
    const _Float16* ga = A  + (size_t)(m0 + rowS) * lda + kk + chS * 16;
    const _Float16* gb = Bm + (size_t)(n0 + rowS) * ldb + kk + chS * 16;
    const uint32_t la = ldsA0 + (uint32_t)buf * TILE_BYTES + sOff;
    const uint32_t lb = ldsB0 + (uint32_t)buf * TILE_BYTES + sOff;
    async_b128(la,      ga);
    async_b128(la + 16, ga + 8);
    async_b128(lb,      gb);
    async_b128(lb + 16, gb + 8);
  };

  stage(0, 0);

  for (int it = 0; it < nIter; ++it) {
    const int cur = it & 1;
    wait_async0();            // our own async fills for 'cur' have landed
    __syncthreads();          // everyone's fills landed; prev reads of 'nxt' done
    if (it + 1 < nIter) stage(cur ^ 1, (it + 1) * 32);  // overlaps compute below

    v16h af[4], bf[2];
#pragma unroll
    for (int tm = 0; tm < 4; ++tm) {
      const _Float16* pa = &As[cur][(wr * 64 + tm * 16 + lm) * RS + 8 * lh];
      ((int4*)&af[tm])[0] = *(const int4*)pa;          // K = 8h .. 8h+7
      ((int4*)&af[tm])[1] = *(const int4*)(pa + 16);   // K = 16+8h ..
    }
#pragma unroll
    for (int tn = 0; tn < 2; ++tn) {
      const _Float16* pb = &Bs[cur][(wc * 32 + tn * 16 + lm) * RS + 8 * lh];
      ((int4*)&bf[tn])[0] = *(const int4*)pb;          // K = 8h .. 8h+7
      ((int4*)&bf[tn])[1] = *(const int4*)(pb + 16);   // K = 16+8h ..
    }

#pragma unroll
    for (int tm = 0; tm < 4; ++tm)
#pragma unroll
      for (int tn = 0; tn < 2; ++tn)
        acc[tm][tn] = __builtin_amdgcn_wmma_f32_16x16x32_f16(
            false, af[tm], false, bf[tn], (short)0, acc[tm][tn], false, false);
  }

  // store: C layout lane -> n = lm, vgpr r -> m = r + 8*lh
#pragma unroll
  for (int tm = 0; tm < 4; ++tm) {
#pragma unroll
    for (int tn = 0; tn < 2; ++tn) {
      const int n = n0 + wc * 32 + tn * 16 + lm;
#pragma unroll
      for (int r = 0; r < 8; ++r) {
        const int m = m0 + wr * 64 + tm * 16 + 8 * lh + r;
        const float v = acc[tm][tn][r];
        if (Ch) Ch[(size_t)m * ldc + n] = (_Float16)v;
        else    Cf[(size_t)m * ldc + n] = v;
      }
    }
  }
}

// ---------------------------------------------------------------------------
// Causal softmax over one score row; writes f16 attn (zeros beyond diagonal).
// ---------------------------------------------------------------------------
__global__ __launch_bounds__(256)
void softmax_causal_kernel(const float* __restrict__ Sc,
                           _Float16* __restrict__ P, int S, float scale) {
  const int q = blockIdx.x;
  const float* row = Sc + (size_t)q * S;
  _Float16* prow = P + (size_t)q * S;
  const int t = threadIdx.x;
  const int n = q + 1;
  __shared__ float red[256];

  float mx = -INFINITY;
  for (int i = t; i < n; i += 256) mx = fmaxf(mx, row[i] * scale);
  red[t] = mx;
  __syncthreads();
  for (int s = 128; s > 0; s >>= 1) {
    if (t < s) red[t] = fmaxf(red[t], red[t + s]);
    __syncthreads();
  }
  mx = red[0];
  __syncthreads();

  float sum = 0.f;
  for (int i = t; i < n; i += 256) sum += __expf(row[i] * scale - mx);
  red[t] = sum;
  __syncthreads();
  for (int s = 128; s > 0; s >>= 1) {
    if (t < s) red[t] += red[t + s];
    __syncthreads();
  }
  const float inv = 1.0f / red[0];

  for (int i = t; i < S; i += 256) {
    float v = (i < n) ? __expf(row[i] * scale - mx) * inv : 0.0f;
    prow[i] = (_Float16)v;
  }
}

// ---------------------------------------------------------------------------
// Host-side launch sequence
// ---------------------------------------------------------------------------
extern "C" void kernel_launch(void* const* d_in, const int* in_sizes, int n_in,
                              void* d_out, int out_size, void* d_ws, size_t ws_size,
                              hipStream_t stream) {
  (void)in_sizes; (void)n_in; (void)out_size; (void)ws_size;
  const float* Wk = (const float*)d_in[0];
  const float* Wq = (const float*)d_in[1];
  const float* Wv = (const float*)d_in[2];
  const float* Wo = (const float*)d_in[3];
  const float* x  = (const float*)d_in[4];
  float* out = (float*)d_out;

  const int B = 4, S = 2048, D = 1024;
  const int MS = B * S;  // 8192

  char* p = (char*)d_ws;
  auto carve = [&](size_t bytes) { char* r = p; p += (bytes + 255) & ~(size_t)255; return r; };
  _Float16* xh    = (_Float16*)carve((size_t)MS * D * 2);
  _Float16* kh    = (_Float16*)carve((size_t)D * D * 2);
  _Float16* qh    = (_Float16*)carve((size_t)D * D * 2);
  _Float16* vh    = (_Float16*)carve((size_t)D * D * 2);
  _Float16* ohT   = (_Float16*)carve((size_t)D * D * 2);   // Wo transposed
  _Float16* Qh    = (_Float16*)carve((size_t)MS * D * 2);
  _Float16* Kh    = (_Float16*)carve((size_t)MS * D * 2);
  _Float16* VTh   = (_Float16*)carve((size_t)MS * D * 2);  // V^T per batch [D,S]
  _Float16* attnh = (_Float16*)carve((size_t)S * S * 2);
  _Float16* Yh    = (_Float16*)carve((size_t)MS * D * 2);
  float*    sc    = (float*)   carve((size_t)S * S * 4);

  dim3 blk(256);

  // 1) convert / transpose inputs to f16
  {
    int n = MS * D;
    f32_to_f16_kernel<<<(n + 255) / 256, 256, 0, stream>>>(x, xh, n);
    n = D * D;
    f32_to_f16_kernel<<<(n + 255) / 256, 256, 0, stream>>>(Wk, kh, n);
    f32_to_f16_kernel<<<(n + 255) / 256, 256, 0, stream>>>(Wq, qh, n);
    f32_to_f16_kernel<<<(n + 255) / 256, 256, 0, stream>>>(Wv, vh, n);
    dim3 gt(D / 256, D);
    transpose_f32_to_f16_kernel<<<gt, 256, 0, stream>>>(Wo, ohT, D, D);
  }

  // 2) Q = x*Wq^T, K = x*Wk^T   (f16 outputs)
  {
    dim3 g(D / 128, MS / 128);
    gemm_wmma_kernel<<<g, blk, 0, stream>>>(xh, D, qh, D, nullptr, Qh, D, D, 0, 0);
    gemm_wmma_kernel<<<g, blk, 0, stream>>>(xh, D, kh, D, nullptr, Kh, D, D, 0, 0);
  }

  // 3) V^T per batch:  VT[d,s] = sum_k x[s,k]*Wv[d,k]  (A=Wv, B=x_b)
  for (int b = 0; b < B; ++b) {
    dim3 g(S / 128, D / 128);
    gemm_wmma_kernel<<<g, blk, 0, stream>>>(vh, D, xh + (size_t)b * S * D, D,
                                            nullptr, VTh + (size_t)b * D * S, S,
                                            D, 0, 0);
  }

  // 4) attention per batch (score buffer reused)
  const float scale = 1.0f / 32.0f;  // 1/sqrt(1024)
  for (int b = 0; b < B; ++b) {
    const _Float16* Qb  = Qh  + (size_t)b * S * D;
    const _Float16* Kb  = Kh  + (size_t)b * S * D;
    const _Float16* VTb = VTh + (size_t)b * D * S;
    _Float16* Yb = Yh + (size_t)b * S * D;

    dim3 g2(S / 128, S / 128);   // scores = Q*K^T (upper-triangle tiles skipped)
    gemm_wmma_kernel<<<g2, blk, 0, stream>>>(Qb, D, Kb, D, sc, nullptr, S, D, 1, 0);

    softmax_causal_kernel<<<S, blk, 0, stream>>>(sc, attnh, S, scale);

    dim3 g3(D / 128, S / 128);   // Y = attn * (VT)^T  (K loop trimmed by causality)
    gemm_wmma_kernel<<<g3, blk, 0, stream>>>(attnh, S, VTb, S, nullptr, Yb, D, S, 0, 1);
  }

  // 5) out = Y * Wo = Y * (WoT)^T   (f32 output)
  {
    dim3 g(D / 128, MS / 128);
    gemm_wmma_kernel<<<g, blk, 0, stream>>>(Yh, D, ohT, D, out, nullptr, D, D, 0, 0);
  }
}
